// Decoder_attri_z_19971597926635
// MI455X (gfx1250) — compile-verified, matrix operand order checked by reading, not run on hardware
//
#include <hip/hip_runtime.h>
#include <cstdint>

#define KOFF 27

typedef __attribute__((ext_vector_type(16))) __bf16 bf16x16;
typedef __attribute__((ext_vector_type(8)))  __bf16 bf16x8;
typedef __attribute__((ext_vector_type(8)))  float  f32x8;

__device__ __forceinline__ unsigned int pk_bf16(float a, float b) {
  union { __bf16 h[2]; unsigned int u; } r;
  r.h[0] = (__bf16)a;
  r.h[1] = (__bf16)b;
  return r.u;
}

__device__ __forceinline__ unsigned short to_bf16(float a) {
  union { __bf16 h; unsigned short u; } r;
  r.h = (__bf16)a;
  return r.u;
}

__device__ __forceinline__ bf16x16 load_frag(const unsigned short* p) {
  // VGPRs 0..3 <- 8 consecutive bf16, VGPRs 4..7 <- 8 consecutive bf16 at +16 elems
  bf16x8 lo = *(const bf16x8*)(p);
  bf16x8 hi = *(const bf16x8*)(p + 16);
  return __builtin_shufflevector(lo, hi, 0,1,2,3,4,5,6,7,8,9,10,11,12,13,14,15);
}

// Async copy of 16 bytes from global (bf16 data) straight into LDS, no VGPR staging.
__device__ __forceinline__ void async_ld16(unsigned lds_addr, const unsigned short* gaddr) {
  asm volatile("global_load_async_to_lds_b128 %0, %1, off"
               :: "v"(lds_addr), "v"(gaddr) : "memory");
}
__device__ __forceinline__ void async_wait0() {
  asm volatile("s_wait_asynccnt 0x0" ::: "memory");
}

// Gather-GEMM-scatter sparse conv: out[out_idx[k][m]] += featsb[in_idx[k][m]] @ Wt[k]
// featsb: bf16 [N,128]; Wt: bf16 [K][COUT][128] (pre-transposed). f32 accumulation.
template<int COUT, int TPB>
__launch_bounds__(128)
__global__ void spconv_wmma(const unsigned short* __restrict__ featsb,
                            const int*            __restrict__ in_idx,
                            const int*            __restrict__ out_idx,
                            const unsigned short* __restrict__ Wt,
                            float*                __restrict__ out,
                            int M)
{
  constexpr int CIN = 128;
  constexpr int NT  = COUT / 16;
  constexpr int AS  = CIN + 8;                 // padded LDS row stride (bf16 elems)
  __shared__ alignas(16) unsigned short sW[COUT * AS];     // W_k^T as [n][c] bf16
  __shared__ alignas(16) unsigned short sA[4][16 * AS];    // per-wave gathered tile
  __shared__ int sO[4][16];                                // per-wave scatter rows

  const int tid  = threadIdx.x;
  const int lane = tid & 31;
  const int wid  = tid >> 5;
  const int k    = blockIdx.x;

  // Async copy of W_k^T (bf16 [COUT][128]) into padded LDS rows.
  {
    unsigned ldsW = (unsigned)(size_t)(void*)sW;
    #pragma unroll
    for (int e = 0; e < COUT * 16; e += 128) {           // 16B chunks
      int ee = e + tid;
      int n  = ee >> 4;
      int c8 = (ee & 15) * 8;
      async_ld16(ldsW + (unsigned)(n * AS + c8) * 2,
                 Wt + ((size_t)k * COUT + n) * CIN + c8);
    }
    async_wait0();
  }
  __syncthreads();

  const int grp = lane >> 4;        // half-wave group (ISA 16-bit WMMA layout)
  const int rlo = lane & 15;        // A-row / B&D-column within tile
  const int tiles = M >> 4;
  int tend = (blockIdx.y + 1) * TPB;
  if (tend > tiles) tend = tiles;
  unsigned short* sAw = sA[wid];
  int* sOw = sO[wid];
  const unsigned ldsA = (unsigned)(size_t)(void*)sAw;

  for (int t = blockIdx.y * TPB + wid; t < tend; t += 4) {
    const int m0 = t * 16;

    // Stage scatter indices once per tile (lanes 0..15).
    if (grp == 0) sOw[rlo] = out_idx[(size_t)k * M + m0 + rlo];

    // Async-gather 16 bf16 feature rows into LDS: 2 rows per instruction
    // (lanes 0-15 -> row rr, lanes 16-31 -> row rr+1), 16B per lane.
    #pragma unroll
    for (int rr = 0; rr < 16; rr += 2) {
      int r   = rr + grp;
      int src = in_idx[(size_t)k * M + m0 + r];
      async_ld16(ldsA + (unsigned)(r * AS + rlo * 8) * 2,
                 featsb + (size_t)src * CIN + rlo * 8);
    }
    async_wait0();

    // D[16,COUT] = A[16,128] x W_k[128,COUT]
    f32x8 acc[NT] = {};
    #pragma unroll
    for (int kt = 0; kt < CIN / 32; ++kt) {
      const unsigned short* pa = &sAw[rlo * AS + kt * 32 + 8 * grp];
      bf16x16 a = load_frag(pa);
      #pragma unroll
      for (int nt = 0; nt < NT; ++nt) {
        const unsigned short* pb = &sW[(nt * 16 + rlo) * AS + kt * 32 + 8 * grp];
        bf16x16 b = load_frag(pb);
        acc[nt] = __builtin_amdgcn_wmma_f32_16x16x32_bf16(
            false, a, false, b, (short)0, acc[nt], false, false);
      }
    }

    // Scatter-add. D layout: VGPR v -> row M = v + 8*grp, col N = nt*16 + rlo.
    #pragma unroll
    for (int v = 0; v < 8; ++v) {
      int grow = sOw[v + 8 * grp];
      float* orow = out + (size_t)grow * COUT + rlo;
      #pragma unroll
      for (int nt = 0; nt < NT; ++nt) {
        __hip_atomic_fetch_add(orow + nt * 16, acc[nt][v],
                               __ATOMIC_RELAXED, __HIP_MEMORY_SCOPE_AGENT);
      }
    }
  }
}

// Final 64 -> 3 layer: too skinny for WMMA; scalar dot per kernel-map pair.
__global__ void spconv_small(const float* __restrict__ feats,
                             const int*   __restrict__ in_idx,
                             const int*   __restrict__ out_idx,
                             const float* __restrict__ W,
                             float*       __restrict__ out,
                             int M)
{
  int p = blockIdx.x * blockDim.x + threadIdx.x;
  if (p >= KOFF * M) return;
  int k = p / M;
  int src = in_idx[p];
  int dst = out_idx[p];
  const float* f = feats + (size_t)src * 64;
  const float* w = W + (size_t)k * 64 * 3;
  float a0 = 0.f, a1 = 0.f, a2 = 0.f;
  #pragma unroll
  for (int c = 0; c < 64; ++c) {
    float x = f[c];
    a0 = fmaf(x, w[c * 3 + 0], a0);
    a1 = fmaf(x, w[c * 3 + 1], a1);
    a2 = fmaf(x, w[c * 3 + 2], a2);
  }
  float* o = out + (size_t)dst * 3;
  __hip_atomic_fetch_add(o + 0, a0, __ATOMIC_RELAXED, __HIP_MEMORY_SCOPE_AGENT);
  __hip_atomic_fetch_add(o + 1, a1, __ATOMIC_RELAXED, __HIP_MEMORY_SCOPE_AGENT);
  __hip_atomic_fetch_add(o + 2, a2, __ATOMIC_RELAXED, __HIP_MEMORY_SCOPE_AGENT);
}

// W[k][c][n] (f32) -> Wt[k][n][c] (bf16)
template<int COUT>
__global__ void cvt_w(const float* __restrict__ W, unsigned short* __restrict__ Wt)
{
  int e = blockIdx.x * blockDim.x + threadIdx.x;
  if (e >= KOFF * 128 * COUT) return;
  int k = e / (128 * COUT);
  int r = e - k * 128 * COUT;
  int c = r / COUT;
  int n = r - c * COUT;
  Wt[((size_t)k * COUT + n) * 128 + c] = to_bf16(W[e]);
}

// f32 -> bf16 mirror, 4 elements/thread
__global__ void cvt_feat(const float* __restrict__ src, unsigned short* __restrict__ dst,
                         int nq)
{
  int i = blockIdx.x * blockDim.x + threadIdx.x;
  if (i >= nq) return;
  const float4 v = ((const float4*)src)[i];
  ((uint2*)dst)[i] = make_uint2(pk_bf16(v.x, v.y), pk_bf16(v.z, v.w));
}

// in-place relu + bf16 mirror
__global__ void relu_cvt(float* __restrict__ x, unsigned short* __restrict__ dst, int nq)
{
  int i = blockIdx.x * blockDim.x + threadIdx.x;
  if (i >= nq) return;
  float4 v = ((float4*)x)[i];
  v.x = fmaxf(v.x, 0.f); v.y = fmaxf(v.y, 0.f);
  v.z = fmaxf(v.z, 0.f); v.w = fmaxf(v.w, 0.f);
  ((float4*)x)[i] = v;
  ((uint2*)dst)[i] = make_uint2(pk_bf16(v.x, v.y), pk_bf16(v.z, v.w));
}

// out[i] = b[i % cout]  (segment_sum + b semantics: bias exactly once per row)
__global__ void bias_init(float* __restrict__ out, const float* __restrict__ b,
                          long long n, int cout)
{
  long long i = (long long)blockIdx.x * blockDim.x + threadIdx.x;
  if (i < n) out[i] = b[(int)(i % cout)];
}

__global__ void relu_ip(float* __restrict__ x, long long n)
{
  long long i = (long long)blockIdx.x * blockDim.x + threadIdx.x;
  if (i < n) x[i] = fmaxf(x[i], 0.0f);
}

extern "C" void kernel_launch(void* const* d_in, const int* in_sizes, int n_in,
                              void* d_out, int out_size, void* d_ws, size_t ws_size,
                              hipStream_t stream)
{
  (void)n_in; (void)out_size; (void)ws_size;
  const float* x     = (const float*)d_in[0];
  const float* W_up0 = (const float*)d_in[1];
  const float* b_up0 = (const float*)d_in[2];
  const float* W_c0  = (const float*)d_in[3];
  const float* b_c0  = (const float*)d_in[4];
  const float* W_up1 = (const float*)d_in[5];
  const float* b_up1 = (const float*)d_in[6];
  const float* W_c1  = (const float*)d_in[7];
  const float* b_c1  = (const float*)d_in[8];
  const float* W_c2  = (const float*)d_in[9];
  const float* b_c2  = (const float*)d_in[10];
  const int* up0_in  = (const int*)d_in[11];
  const int* up0_out = (const int*)d_in[12];
  const int* c0_in   = (const int*)d_in[13];
  const int* c0_out  = (const int*)d_in[14];
  const int* up1_in  = (const int*)d_in[15];
  const int* up1_out = (const int*)d_in[16];
  const int* c1_in   = (const int*)d_in[17];
  const int* c1_out  = (const int*)d_in[18];
  const int* c2_in   = (const int*)d_in[19];
  const int* c2_out  = (const int*)d_in[20];

  const int N0 = 30000, N1 = 100000, N2 = 300000;
  const int M0 = in_sizes[11] / KOFF;   // 40000
  const int M1 = in_sizes[15] / KOFF;   // 50000
  const int M2 = in_sizes[19] / KOFF;   // 80000

  // d_out: [out (N2*3)] [out_cls_1 (N2*64)] [out_cls_0 (N1*128)]
  float* out0 = (float*)d_out;
  float* oc1  = out0 + (size_t)N2 * 3;
  float* oc0  = oc1 + (size_t)N2 * 64;

  // d_ws: [f32 accum h (N2*128)] [bf16 feature mirror (N2*128)] [bf16 weights]
  float* h = (float*)d_ws;
  unsigned short* fb = (unsigned short*)(h + (size_t)N2 * 128);
  unsigned short* wt_up0 = fb + (size_t)N2 * 128;
  unsigned short* wt_c0  = wt_up0 + (size_t)KOFF * 128 * 128;
  unsigned short* wt_up1 = wt_c0  + (size_t)KOFF * 128 * 128;
  unsigned short* wt_c1  = wt_up1 + (size_t)KOFF * 128 * 128;

  constexpr int TPB = 32;               // m-tiles per block (8 per wave)
  auto blks = [](long long n) { return (unsigned)((n + 255) / 256); };
  dim3 g0(KOFF, (unsigned)((M0 / 16 + TPB - 1) / TPB));
  dim3 g1(KOFF, (unsigned)((M1 / 16 + TPB - 1) / TPB));

  // Pre-convert weights: f32 [K][128][COUT] -> bf16 [K][COUT][128]
  cvt_w<128><<<blks(KOFF * 128 * 128), 256, 0, stream>>>(W_up0, wt_up0);
  cvt_w<128><<<blks(KOFF * 128 * 128), 256, 0, stream>>>(W_c0,  wt_c0);
  cvt_w<128><<<blks(KOFF * 128 * 128), 256, 0, stream>>>(W_up1, wt_up1);
  cvt_w<64><<<blks(KOFF * 128 * 64),  256, 0, stream>>>(W_c1,  wt_c1);

  // h1 = up0(x)
  cvt_feat<<<blks(N0 * 32), 256, 0, stream>>>(x, fb, N0 * 32);
  bias_init<<<blks((long long)N1 * 128), 256, 0, stream>>>(h, b_up0, (long long)N1 * 128, 128);
  spconv_wmma<128, TPB><<<g0, 128, 0, stream>>>(fb, up0_in, up0_out, wt_up0, h, M0);

  // out_cls_0 = relu(conv0(h1))
  cvt_feat<<<blks(N1 * 32), 256, 0, stream>>>(h, fb, N1 * 32);
  bias_init<<<blks((long long)N1 * 128), 256, 0, stream>>>(oc0, b_c0, (long long)N1 * 128, 128);
  spconv_wmma<128, TPB><<<g0, 128, 0, stream>>>(fb, c0_in, c0_out, wt_c0, oc0, M0);
  relu_cvt<<<blks(N1 * 32), 256, 0, stream>>>(oc0, fb, N1 * 32);

  // h2 = up1(out_cls_0)
  bias_init<<<blks((long long)N2 * 128), 256, 0, stream>>>(h, b_up1, (long long)N2 * 128, 128);
  spconv_wmma<128, TPB><<<g1, 128, 0, stream>>>(fb, up1_in, up1_out, wt_up1, h, M1);

  // out_cls_1 = relu(conv1(h2))
  cvt_feat<<<blks((long long)N2 * 32), 256, 0, stream>>>(h, fb, N2 * 32);
  bias_init<<<blks((long long)N2 * 64), 256, 0, stream>>>(oc1, b_c1, (long long)N2 * 64, 64);
  spconv_wmma<64, TPB><<<g1, 128, 0, stream>>>(fb, c1_in, c1_out, wt_c1, oc1, M1);
  relu_ip<<<blks((long long)N2 * 64), 256, 0, stream>>>(oc1, (long long)N2 * 64);

  // out = conv2(out_cls_1)
  bias_init<<<blks((long long)N2 * 3), 256, 0, stream>>>(out0, b_c2, (long long)N2 * 3, 3);
  spconv_small<<<(KOFF * M2 + 255) / 256, 256, 0, stream>>>(oc1, c2_in, c2_out, W_c2, out0, M2);
}
